// Gemma4AudioAttention_33646773797298
// MI455X (gfx1250) — matrix operand order repaired and checked by reading, not compiled, not run on hardware
//
#include <hip/hip_runtime.h>
#include <cmath>

#define B_    2
#define S_    8192
#define HID_  1024
#define H_    8
#define D_    128
#define CHUNK_ 128
#define PAST_  127
#define CTX_   383
#define POS_   384
#define NB_    64
#define HD_    1024
#define BS_    16384
#define CAP_   50.0f
#define LPAD_  7            /* left zero-pad of vT rows: makes indices 8-aligned */
#define SP2_   (S_ + 16)    /* padded vT row length */
#define KK_    1024         /* K dim of every GEMM in this problem */

typedef __bf16 bf16_t;
typedef __attribute__((ext_vector_type(16))) __bf16 v16bf;
typedef __attribute__((ext_vector_type(8)))  float  v8f;
typedef __attribute__((ext_vector_type(4)))  unsigned uint4v;  // 16B, align 16

union Frag { v16bf v; unsigned u[8]; uint4v q[2]; };

__device__ __forceinline__ v8f zero8() {
  v8f z = {0.f,0.f,0.f,0.f,0.f,0.f,0.f,0.f};
  return z;
}

// A-matrix (16x32 bf16) K offset for VGPR j, lane-half h (ISA 7.12.2 table)
__device__ __forceinline__ int kofA(int j, int half) {
  return ((j < 4) ? (2 * j) : (16 + 2 * (j - 4))) + (half ? 8 : 0);
}
// B-operand (32x16 bf16) K offset in this code's contiguous order (k = 16*half + 2j)
__device__ __forceinline__ int kofB(int j, int half) {
  return 2 * j + (half ? 16 : 0);
}

// A fragment = 32 contiguous bytes at +32*half; must be 16B aligned
__device__ __forceinline__ void ld_a16(Frag& f, const bf16_t* p, int half) {
  const char* c = (const char*)p + (half ? 32 : 0);
  f.q[0] = *(const uint4v*)c;
  f.q[1] = *(const uint4v*)(c + 16);
}
// B fragment = 32 contiguous bytes at +32*half; must be 16B aligned
__device__ __forceinline__ void ld_b16(Frag& f, const bf16_t* p, int half) {
  const char* c = (const char*)p + (half ? 32 : 0);
  f.q[0] = *(const uint4v*)c;
  f.q[1] = *(const uint4v*)(c + 16);
}

__device__ __forceinline__ v8f wmma_bf16(const Frag& a, const Frag& b, v8f c) {
  return __builtin_amdgcn_wmma_f32_16x16x32_bf16(false, a.v, false, b.v,
                                                 (short)0, c, false, false);
}

// ---------------------------------------------------------------------------
// elementwise converters
// ---------------------------------------------------------------------------
__global__ void k_f32_to_bf16(const float* __restrict__ x, bf16_t* __restrict__ y, long n) {
  long i = (long)blockIdx.x * blockDim.x + threadIdx.x;
  long stride = (long)gridDim.x * blockDim.x;
  for (; i < n; i += stride) y[i] = (bf16_t)x[i];
}

// x: K x N row-major f32 ; y: N x K row-major bf16 (transposed)
__global__ void k_f32_to_bf16_T(const float* __restrict__ x, bf16_t* __restrict__ y,
                                int K, int N) {
  long total = (long)K * N;
  long i = (long)blockIdx.x * blockDim.x + threadIdx.x;
  long stride = (long)gridDim.x * blockDim.x;
  for (; i < total; i += stride) {
    long nn = i / K, kk = i - nn * K;
    y[i] = (bf16_t)x[kk * (long)N + nn];
  }
}

__global__ void k_softplus(const float* __restrict__ pds, float* __restrict__ out, float qbase) {
  int d = threadIdx.x;
  if (d < D_) out[d] = qbase * log1pf(__expf(pds[d]));
}

// zero the pad elements of every padded vT row
__global__ void k_vpad(bf16_t* __restrict__ vtb) {
  int i = blockIdx.x * blockDim.x + threadIdx.x;
  if (i < B_ * H_ * D_) {
    bf16_t* row = vtb + (long)i * SP2_;
#pragma unroll
    for (int j = 0; j < LPAD_; ++j) row[j] = (bf16_t)0.f;
#pragma unroll
    for (int j = 0; j < 9; ++j) row[S_ + LPAD_ + j] = (bf16_t)0.f;
  }
}

// ---------------------------------------------------------------------------
// generic bf16 WMMA GEMM (K fixed at 1024): C = A(Mx1024) * Bt^T (Bt Nx1024).
// Block tile 128x64, 8 waves; each wave one 16x64 strip. K loop fully
// unrolled: every fragment load is a b128 with an immediate offset, scheduler
// pipelines loads across the 4 wmma per step. Epilogue scatters per `mode`.
// ---------------------------------------------------------------------------
enum { MODE_Q = 0, MODE_K = 1, MODE_VT = 2, MODE_REL = 3, MODE_OUTF32 = 4 };

__global__ __launch_bounds__(256) void wmma_gemm(
    const bf16_t* __restrict__ A, const bf16_t* __restrict__ Bt,
    int N, int mode, float kscale,
    const float* __restrict__ qscale, void* __restrict__ out)
{
  const int tid = threadIdx.x, lane = tid & 31, w = tid >> 5;
  const int r = lane & 15, half = lane >> 4;
  const long rowbase = (long)blockIdx.y * 128 + w * 16;
  const long colbase = (long)blockIdx.x * 64;

  v8f acc[4];
#pragma unroll
  for (int c = 0; c < 4; ++c) acc[c] = zero8();

  const bf16_t* arow = A + (rowbase + r) * (long)KK_;
  const bf16_t* brow0 = Bt + (colbase + r) * (long)KK_;
  const bf16_t* brow1 = brow0 + 16 * KK_;
  const bf16_t* brow2 = brow0 + 32 * KK_;
  const bf16_t* brow3 = brow0 + 48 * KK_;

#pragma unroll
  for (int s = 0; s < KK_ / 32; ++s) {
    Frag af, b0, b1, b2, b3;
    ld_a16(af, arow + s * 32, half);
    ld_b16(b0, brow0 + s * 32, half);
    ld_b16(b1, brow1 + s * 32, half);
    ld_b16(b2, brow2 + s * 32, half);
    ld_b16(b3, brow3 + s * 32, half);
    acc[0] = wmma_bf16(af, b0, acc[0]);
    acc[1] = wmma_bf16(af, b1, acc[1]);
    acc[2] = wmma_bf16(af, b2, acc[2]);
    acc[3] = wmma_bf16(af, b3, acc[3]);
  }

#pragma unroll
  for (int t = 0; t < 4; ++t) {
#pragma unroll
    for (int e = 0; e < 8; ++e) {
      long row = rowbase + e + 8 * half;
      long col = colbase + t * 16 + r;
      float v = acc[t][e];
      if (mode == MODE_Q || mode == MODE_K) {
        long b = row >> 13, s = row & (S_ - 1);
        long h = col >> 7,  d = col & (D_ - 1);
        v *= (mode == MODE_Q) ? qscale[d] : kscale;
        ((bf16_t*)out)[(((b * H_ + h) * S_ + s) << 7) + d] = (bf16_t)v;
      } else if (mode == MODE_VT) {
        long b = row >> 13, s = row & (S_ - 1);
        long h = col >> 7,  d = col & (D_ - 1);
        ((bf16_t*)out)[((b * H_ + h) * D_ + d) * (long)SP2_ + s + LPAD_] = (bf16_t)v;
      } else if (mode == MODE_REL) {
        long h = col >> 7, d = col & (D_ - 1);
        ((bf16_t*)out)[(h * POS_ + row) * D_ + d] = (bf16_t)v;
      } else {
        ((float*)out)[row * (long)N + col] = v;
      }
    }
  }
}

// ---------------------------------------------------------------------------
// attention kernel: one block per (b,h,chunk). 8 waves, wave w owns rows
// [16w,16w+16). Phase 1: bd_orig (128x384) -> LDS bf16. Phase 2: streaming
// over 12 x 32 ctx columns: QK wmma, add skewed bd, tanh-cap, exp(l-50),
// P*V wmma accumulation; final 1/sum normalization.
// ---------------------------------------------------------------------------
__global__ __launch_bounds__(256) void attn_kernel(
    const bf16_t* __restrict__ q, const bf16_t* __restrict__ k,
    const bf16_t* __restrict__ vT, const bf16_t* __restrict__ relk,
    bf16_t* __restrict__ ctx_out)
{
  __shared__ __align__(16) bf16_t BD[128 * 384];   // bd_orig, row stride POS=384
  __shared__ __align__(16) bf16_t PS[8][16 * 32];  // per-wave prob tiles

  const int tid = threadIdx.x, lane = tid & 31, w = tid >> 5;
  const int r = lane & 15, half = lane >> 4;
  const int bid = blockIdx.x;
  const int n = bid % NB_;
  const int h = (bid / NB_) % H_;
  const int b = bid / (NB_ * H_);

  const bf16_t* qh   = q    + ((long)(b * H_ + h) * S_) * D_;
  const bf16_t* kh   = k    + ((long)(b * H_ + h) * S_) * D_;
  const bf16_t* vth  = vT   + ((long)(b * H_ + h) * D_) * (long)SP2_;
  const bf16_t* relh = relk + (long)h * POS_ * D_;

  const int  s0    = n * CHUNK_;
  const long wbase = (long)s0 - PAST_;  // ctx j -> global s = wbase + j

  // resident Q A-fragments for this wave's 16-row strip (K = D = 128, 4 slices)
  Frag qa[4];
  {
    const bf16_t* row = qh + (long)(s0 + 16 * w + r) * D_;
#pragma unroll
    for (int kk = 0; kk < 4; ++kk) ld_a16(qa[kk], row + kk * 32, half);
  }

  // ---- Phase 1: bd_orig rows [16w,16w+16) x POS=384 -> BD (bf16) ----
  for (int nt = 0; nt < 24; ++nt) {
    v8f acc = zero8();
    const bf16_t* prow = relh + (long)(nt * 16 + r) * D_;
#pragma unroll
    for (int kk = 0; kk < 4; ++kk) {
      Frag bfrag;
      ld_b16(bfrag, prow + kk * 32, half);
      acc = wmma_bf16(qa[kk], bfrag, acc);
    }
#pragma unroll
    for (int e = 0; e < 8; ++e) {
      int row = 16 * w + e + 8 * half;
      BD[row * 384 + nt * 16 + r] = (bf16_t)acc[e];
    }
  }
  __syncthreads();

  // ---- Phase 2: streaming softmax-free attention (cap bounds logits) ----
  v8f o[8];
#pragma unroll
  for (int dt = 0; dt < 8; ++dt) o[dt] = zero8();
  float ssum[8] = {0.f,0.f,0.f,0.f,0.f,0.f,0.f,0.f};

  for (int nstep = 0; nstep < 12; ++nstep) {
    const int cb = nstep * 32;
    // whole 32-wide ctx slice maps to in-range source rows?
    const bool interior = (wbase + cb >= 0) && (wbase + cb + 32 <= (long)S_);

    // QK^T for two 16-wide ctx tiles (zero-fill out-of-range k rows: the
    // reference zero-pads k/v and softmaxes over those positions)
    v8f a0 = zero8(), a1 = zero8();
    {
      long sc0 = wbase + cb + r;
      long sc1 = sc0 + 16;
      const bf16_t* kr0 = kh + sc0 * D_;
      const bf16_t* kr1 = kh + sc1 * D_;
      if (interior) {
#pragma unroll
        for (int kk = 0; kk < 4; ++kk) {
          Frag f0, f1;
          ld_b16(f0, kr0 + kk * 32, half);
          ld_b16(f1, kr1 + kk * 32, half);
          a0 = wmma_bf16(qa[kk], f0, a0);
          a1 = wmma_bf16(qa[kk], f1, a1);
        }
      } else {
        bool v0 = (sc0 >= 0 && sc0 < S_);
        bool v1 = (sc1 >= 0 && sc1 < S_);
#pragma unroll
        for (int kk = 0; kk < 4; ++kk) {
          Frag f0, f1;
#pragma unroll
          for (int j = 0; j < 8; ++j) {
            int kb = kk * 32 + kofB(j, half);
            f0.u[j] = v0 ? *(const unsigned*)(kr0 + kb) : 0u;
            f1.u[j] = v1 ? *(const unsigned*)(kr1 + kb) : 0u;
          }
          a0 = wmma_bf16(qa[kk], f0, a0);
          a1 = wmma_bf16(qa[kk], f1, a1);
        }
      }
    }

    // add skewed bd (BD is linear under flat = c*383 + k), cap, exponentiate
#pragma unroll
    for (int e = 0; e < 8; ++e) {
      int row = 16 * w + e + 8 * half;
      int k0 = cb + r;
      int k1 = k0 + 16;
      float p0, p1;
      if (k0 < CTX_) {
        float x = a0[e] + (float)BD[row * 383 + k0];
        x = CAP_ * tanhf(x * (1.0f / CAP_));
        p0 = __expf(x - CAP_);
      } else p0 = 0.f;
      if (k1 < CTX_) {
        float x = a1[e] + (float)BD[row * 383 + k1];
        x = CAP_ * tanhf(x * (1.0f / CAP_));
        p1 = __expf(x - CAP_);
      } else p1 = 0.f;
      ssum[e] += p0 + p1;
      int rr = e + 8 * half;
      PS[w][rr * 32 + r]      = (bf16_t)p0;
      PS[w][rr * 32 + 16 + r] = (bf16_t)p1;
    }
    asm volatile("s_wait_dscnt 0" ::: "memory");  // same-wave LDS RAW

    // probs as A-fragment (16x32) straight from LDS (16B aligned -> b128)
    Frag pf;
    ld_a16(pf, &PS[w][r * 32], half);

    // P * V : padded vT rows (LPAD_=7 zeros) make the padded index 8-aligned
    // -> two aligned b128 per fragment on interior steps; guarded dwords on
    // boundary steps (pads reproduce the reference's zero-padded v).
#pragma unroll
    for (int dt = 0; dt < 8; ++dt) {
      const bf16_t* vrow = vth + (long)(dt * 16 + r) * SP2_;
      Frag vf;
      if (interior) {
        ld_b16(vf, vrow + (wbase + cb + LPAD_), half);
      } else {
#pragma unroll
        for (int j = 0; j < 8; ++j) {
          long pidx = wbase + cb + kofB(j, half) + LPAD_;  // even
          vf.u[j] = (pidx >= 0 && pidx + 1 < (long)SP2_)
                      ? *(const unsigned*)(vrow + pidx) : 0u;
        }
      }
      o[dt] = wmma_bf16(pf, vf, o[dt]);
    }
  }

  // row sums: butterfly over the 16 lanes holding each row's columns
#pragma unroll
  for (int e = 0; e < 8; ++e) {
    float v = ssum[e];
    v += __shfl_xor(v, 1, 32);
    v += __shfl_xor(v, 2, 32);
    v += __shfl_xor(v, 4, 32);
    v += __shfl_xor(v, 8, 32);
    ssum[e] = 1.0f / v;
  }

#pragma unroll
  for (int dt = 0; dt < 8; ++dt) {
#pragma unroll
    for (int e = 0; e < 8; ++e) {
      int row = 16 * w + e + 8 * half;
      long s = s0 + row;
      long col = (long)h * D_ + dt * 16 + r;
      ctx_out[((long)b * S_ + s) * HD_ + col] = (bf16_t)(o[dt][e] * ssum[e]);
    }
  }
}

// ---------------------------------------------------------------------------
extern "C" void kernel_launch(void* const* d_in, const int* in_sizes, int n_in,
                              void* d_out, int out_size, void* d_ws, size_t ws_size,
                              hipStream_t stream) {
  (void)in_sizes; (void)n_in; (void)out_size; (void)ws_size;
  const float* hs    = (const float*)d_in[0];
  const float* pos   = (const float*)d_in[1];
  const float* Wq    = (const float*)d_in[2];
  const float* Wk    = (const float*)d_in[3];
  const float* Wv    = (const float*)d_in[4];
  const float* Wrel  = (const float*)d_in[5];
  const float* Wpost = (const float*)d_in[6];
  const float* pds   = (const float*)d_in[7];
  float* out = (float*)d_out;

  char* ws = (char*)d_ws;
  size_t cur = 0;
  auto take = [&](size_t bytes) {
    size_t o = cur; cur += (bytes + 255) & ~(size_t)255; return o;
  };
  bf16_t* hsb    = (bf16_t*)(ws + take((size_t)BS_ * HID_ * 2));
  bf16_t* wqT    = (bf16_t*)(ws + take((size_t)HID_ * HD_ * 2));
  bf16_t* wkT    = (bf16_t*)(ws + take((size_t)HID_ * HD_ * 2));
  bf16_t* wvT    = (bf16_t*)(ws + take((size_t)HID_ * HD_ * 2));
  bf16_t* wrelT  = (bf16_t*)(ws + take((size_t)HID_ * HD_ * 2));
  bf16_t* wpostT = (bf16_t*)(ws + take((size_t)HID_ * HID_ * 2));
  bf16_t* posb   = (bf16_t*)(ws + take((size_t)POS_ * HID_ * 2));
  float*  qscale = (float*)(ws + take(512));
  bf16_t* qb     = (bf16_t*)(ws + take((size_t)BS_ * HD_ * 2));
  bf16_t* kb     = (bf16_t*)(ws + take((size_t)BS_ * HD_ * 2));
  bf16_t* vtb    = (bf16_t*)(ws + take((size_t)B_ * H_ * D_ * SP2_ * 2));
  bf16_t* relb   = (bf16_t*)(ws + take((size_t)H_ * POS_ * D_ * 2));
  bf16_t* ctxb   = (bf16_t*)(ws + take((size_t)BS_ * HD_ * 2));

  const int thr = 256;
  long nhs = (long)BS_ * HID_;
  k_f32_to_bf16<<<dim3((unsigned)((nhs + thr - 1) / thr)), thr, 0, stream>>>(hs, hsb, nhs);
  long nw = (long)HID_ * HD_;
  unsigned gw = (unsigned)((nw + thr - 1) / thr);
  k_f32_to_bf16_T<<<gw, thr, 0, stream>>>(Wq,    wqT,    HID_, HD_);
  k_f32_to_bf16_T<<<gw, thr, 0, stream>>>(Wk,    wkT,    HID_, HD_);
  k_f32_to_bf16_T<<<gw, thr, 0, stream>>>(Wv,    wvT,    HID_, HD_);
  k_f32_to_bf16_T<<<gw, thr, 0, stream>>>(Wrel,  wrelT,  HID_, HD_);
  k_f32_to_bf16_T<<<gw, thr, 0, stream>>>(Wpost, wpostT, HID_, HID_);
  long npos = (long)POS_ * HID_;
  k_f32_to_bf16<<<dim3((unsigned)((npos + thr - 1) / thr)), thr, 0, stream>>>(pos, posb, npos);
  k_vpad<<<dim3((B_ * H_ * D_ + thr - 1) / thr), thr, 0, stream>>>(vtb);

  float qbase  = (float)(pow((double)D_, -0.5) / log(2.0));
  float kscale = (float)(log1p(exp(1.0)) / log(2.0));
  k_softplus<<<1, 128, 0, stream>>>(pds, qscale, qbase);

  dim3 gq(HD_ / 64, BS_ / 128);
  wmma_gemm<<<gq, 256, 0, stream>>>(hsb, wqT, HD_, MODE_Q,  0.f,    qscale, qb);
  wmma_gemm<<<gq, 256, 0, stream>>>(hsb, wkT, HD_, MODE_K,  kscale, qscale, kb);
  wmma_gemm<<<gq, 256, 0, stream>>>(hsb, wvT, HD_, MODE_VT, 0.f,    qscale, vtb);
  dim3 gr(HD_ / 64, POS_ / 128);
  wmma_gemm<<<gr, 256, 0, stream>>>(posb, wrelT, HD_, MODE_REL, 0.f, qscale, relb);

  attn_kernel<<<dim3(B_ * H_ * NB_), 256, 0, stream>>>(qb, kb, vtb, relb, ctxb);

  dim3 gf(HID_ / 64, BS_ / 128);
  wmma_gemm<<<gf, 256, 0, stream>>>(ctxb, wpostT, HID_, MODE_OUTF32, 0.f, qscale, out);
}
